// GNN_Model_1_84825604096009
// MI455X (gfx1250) — matrix operand order repaired and checked by reading, not compiled, not run on hardware
//
#include <hip/hip_runtime.h>
#include <stdint.h>

#define NNODES 100000
#define NEDGES 1600000
#define CH 128

typedef __bf16 v16bf __attribute__((ext_vector_type(16)));
typedef float  v8f   __attribute__((ext_vector_type(8)));

struct __attribute__((aligned(16))) U4 { unsigned int a, b, c, d; };
struct Bytes32 { U4 lo, hi; };

__device__ __forceinline__ unsigned short f2bf(float f) {
    unsigned int u = __float_as_uint(f);
    u += 0x7fffu + ((u >> 16) & 1u);   // round-to-nearest-even
    return (unsigned short)(u >> 16);
}

// ---------------------------------------------------------------- utilities
__global__ void k_zero(float* __restrict__ p, int n) {
    int i = blockIdx.x * blockDim.x + threadIdx.x;
    if (i < n) p[i] = 0.f;
}

__global__ void k_deg(const int* __restrict__ ei, float* __restrict__ deg) {
    int e = blockIdx.x * blockDim.x + threadIdx.x;
    if (e < NEDGES) atomicAdd(&deg[ei[NEDGES + e]], 1.0f);
}

__global__ void k_dis(float* __restrict__ deg) {
    int i = blockIdx.x * blockDim.x + threadIdx.x;
    if (i < NNODES) deg[i] = rsqrtf(deg[i] + 1.0f);
}

__global__ void k_f2bf(const float* __restrict__ in, unsigned short* __restrict__ out, int n) {
    int i = blockIdx.x * blockDim.x + threadIdx.x;
    if (i < n) out[i] = f2bf(in[i]);
}

// W is [K=128, Cout] row-major; produce Wt[n*128 + k] (column-major, contiguous K per lane)
__global__ void k_wt_bf(const float* __restrict__ W, unsigned short* __restrict__ Wt, int Cout) {
    int i = blockIdx.x * blockDim.x + threadIdx.x;
    if (i < 128 * Cout) {
        int k = i / Cout, n = i % Cout;
        Wt[n * 128 + k] = f2bf(W[i]);
    }
}

// ------------------------------------------------------------- WMMA GEMM
// C[M x Cout] = A[M x 128](bf16) * Bt[Cout x 128](bf16, col-major of W), f32 accum.
// One wave per 16x16 output tile; blockIdx.x = row tile; wave-in-block = col tile.
// Fragment layouts used here (contiguous K, same order for A and B):
//   A (16x32 bf16): lane l<16 -> row l, K = k0+{0..15}; lane l+16 -> row l, K = k0+{16..31}
//   B (32x16 bf16): lane l<16 -> col l, K = k0+{0..15}; lane l+16 -> col l, K = k0+{16..31}
//   D (16x16 f32):  vgpr j: lanes 0-15 row j, lanes 16-31 row j+8
__global__ void k_gemm_bf16(const unsigned short* __restrict__ A,
                            const unsigned short* __restrict__ Bt,
                            float* __restrict__ C, int Cout) {
    const int lane = threadIdx.x & 31;
    const int wave = threadIdx.x >> 5;
    const int row0 = blockIdx.x * 16;
    const int col0 = wave * 16;
    const int half = lane >> 4;
    const int l16  = lane & 15;

    const unsigned short* arow = A  + (size_t)(row0 + l16) * 128;
    const unsigned short* bcol = Bt + (size_t)(col0 + l16) * 128;

    v8f acc = {0.f, 0.f, 0.f, 0.f, 0.f, 0.f, 0.f, 0.f};
#pragma unroll
    for (int k0 = 0; k0 < 128; k0 += 32) {
        Bytes32 af, bfr;
        af.lo  = *(const U4*)(arow + k0 + half * 16);
        af.hi  = *(const U4*)(arow + k0 + half * 16 + 8);
        const unsigned short* bp = bcol + k0 + half * 16;
        bfr.lo = *(const U4*)(bp);
        bfr.hi = *(const U4*)(bp + 8);
        v16bf a = __builtin_bit_cast(v16bf, af);
        v16bf b = __builtin_bit_cast(v16bf, bfr);
        acc = __builtin_amdgcn_wmma_f32_16x16x32_bf16(false, a, false, b,
                                                      (short)0, acc, false, false);
    }
    float* crow = C + (size_t)(row0 + half * 8) * Cout + col0 + l16;
#pragma unroll
    for (int j = 0; j < 8; ++j) crow[(size_t)j * Cout] = acc[j];
}

// ----------------------------------------------------- edge scatter (wave/edge)
__global__ void k_scatter(const float* __restrict__ lin, const int* __restrict__ ei,
                          const float* __restrict__ dis, float* __restrict__ agg) {
    int t = blockIdx.x * blockDim.x + threadIdx.x;
    int e = t >> 5, lane = t & 31;
    if (e >= NEDGES) return;
    int s = ei[e], d = ei[NEDGES + e];
    float w = dis[s] * dis[d];
    const float* sp = lin + (size_t)s * CH + lane;
    float*       dp = agg + (size_t)d * CH + lane;
#pragma unroll
    for (int c = 0; c < 4; ++c) atomicAdd(dp + 32 * c, sp[32 * c] * w);
}

// agg + lin*self_norm + bias -> relu (+residual)
__global__ void k_combine(const float* __restrict__ agg, const float* __restrict__ lin,
                          const float* __restrict__ dis, const float* __restrict__ bias,
                          const float* __restrict__ resid, float* __restrict__ out) {
    int i = blockIdx.x * blockDim.x + threadIdx.x;
    if (i >= NNODES * CH) return;
    int node = i >> 7, c = i & 127;
    float sn = dis[node]; sn *= sn;
    float v = fmaxf(agg[i] + lin[i] * sn + bias[c], 0.f);
    if (resid) v += resid[i];
    out[i] = v;
}

// in-place: relu(((h + bias) - mean) * gamma*rsqrt(var+eps) + beta)
__global__ void k_bn_relu(float* __restrict__ h, const float* __restrict__ b,
                          const float* __restrict__ g, const float* __restrict__ beta,
                          const float* __restrict__ m, const float* __restrict__ var,
                          int n, int logD) {
    int i = blockIdx.x * blockDim.x + threadIdx.x;
    if (i >= n) return;
    int c = i & ((1 << logD) - 1);
    float scale = g[c] * rsqrtf(var[c] + 1e-5f);
    h[i] = fmaxf((h[i] + b[c] - m[c]) * scale + beta[c], 0.f);
}

// radius = softplus(r1[64] . rW2 + rb2)
__global__ void k_radius(const float* __restrict__ r1, const float* __restrict__ w,
                         const float* __restrict__ b, float* __restrict__ radius) {
    int i = blockIdx.x * blockDim.x + threadIdx.x;
    if (i >= NNODES) return;
    const float* rp = r1 + (size_t)i * 64;
    float acc = b[0];
#pragma unroll
    for (int j = 0; j < 64; ++j) acc += rp[j] * w[j];
    radius[i] = (acc > 0.f) ? acc + log1pf(expf(-acc)) : log1pf(expf(acc));
}

// coords = c2[64] @ pW3[64x2] + pb3; out = coords / max(|coords|,1e-12) * radius
__global__ void k_final(const float* __restrict__ c2, const float* __restrict__ pW3,
                        const float* __restrict__ pb3, const float* __restrict__ radius,
                        float* __restrict__ out) {
    int i = blockIdx.x * blockDim.x + threadIdx.x;
    if (i >= NNODES) return;
    const float* cp = c2 + (size_t)i * 64;
    float cx = pb3[0], cy = pb3[1];
#pragma unroll
    for (int j = 0; j < 64; ++j) {
        float v = cp[j];
        cx += v * pW3[2 * j];
        cy += v * pW3[2 * j + 1];
    }
    float nrm = fmaxf(sqrtf(cx * cx + cy * cy), 1e-12f);
    float s = radius[i] / nrm;
    out[2 * i]     = cx * s;
    out[2 * i + 1] = cy * s;
}

extern "C" void kernel_launch(void* const* d_in, const int* in_sizes, int n_in,
                              void* d_out, int out_size, void* d_ws, size_t ws_size,
                              hipStream_t stream) {
    const float* x   = (const float*)d_in[0];
    const int*   ei  = (const int*)d_in[1];   // [2,E] int32 (src row, dst row)
    const float* W1  = (const float*)d_in[2],  *b1  = (const float*)d_in[3];
    const float* W2  = (const float*)d_in[4],  *b2  = (const float*)d_in[5];
    const float* W3  = (const float*)d_in[6],  *b3  = (const float*)d_in[7];
    const float* pW1 = (const float*)d_in[8],  *pb1 = (const float*)d_in[9];
    const float* pW2 = (const float*)d_in[10], *pb2 = (const float*)d_in[11];
    const float* pW3 = (const float*)d_in[12], *pb3 = (const float*)d_in[13];
    const float* rW1 = (const float*)d_in[14], *rb1 = (const float*)d_in[15];
    const float* rW2 = (const float*)d_in[16], *rb2 = (const float*)d_in[17];
    const float* pbn1g = (const float*)d_in[18], *pbn1b = (const float*)d_in[19],
               * pbn1m = (const float*)d_in[20], *pbn1v = (const float*)d_in[21];
    const float* pbn2g = (const float*)d_in[22], *pbn2b = (const float*)d_in[23],
               * pbn2m = (const float*)d_in[24], *pbn2v = (const float*)d_in[25];
    const float* rbn1g = (const float*)d_in[26], *rbn1b = (const float*)d_in[27],
               * rbn1m = (const float*)d_in[28], *rbn1v = (const float*)d_in[29];

    const size_t N = NNODES;
    float* ws     = (float*)d_ws;
    float* dis    = ws;                 // N
    float* radius = ws + N;             // N
    float* hA     = ws + 2 * N;         // N*128
    float* hB     = hA + N * 128;       // N*128
    float* agg    = hB + N * 128;       // N*128
    float* lin    = agg + N * 128;      // N*128
    unsigned short* hbf = (unsigned short*)(lin + N * 128);  // N*128 bf16
    unsigned short* wbf = hbf + N * 128;                     // 128*128 bf16
    float* rbuf = agg;                  // head reuse: N*64
    float* c2   = agg + N * 64;         // head reuse: N*64

    const int TPB = 256;
    auto cdiv = [](long long a, long long b) { return (int)((a + b - 1) / b); };
    const int nNC  = (int)(N * 128);
    const int gNC  = cdiv(nNC, TPB);
    const int gN   = cdiv((long long)N, TPB);

    // degrees -> dis
    k_zero<<<gN, TPB, 0, stream>>>(dis, (int)N);
    k_deg<<<cdiv(NEDGES, TPB), TPB, 0, stream>>>(ei, dis);
    k_dis<<<gN, TPB, 0, stream>>>(dis);

    // three GCN layers
    const float* lin_in[3]  = { x,  hA, hB };
    const float* lW[3]      = { W1, W2, W3 };
    const float* lb[3]      = { b1, b2, b3 };
    const float* lres[3]    = { nullptr, hA, hB };
    float*       lout[3]    = { hA, hB, hA };
    for (int l = 0; l < 3; ++l) {
        k_f2bf<<<gNC, TPB, 0, stream>>>(lin_in[l], hbf, nNC);
        k_wt_bf<<<cdiv(128 * 128, TPB), TPB, 0, stream>>>(lW[l], wbf, 128);
        k_gemm_bf16<<<dim3((int)(N / 16)), dim3(8 * 32), 0, stream>>>(hbf, wbf, lin, 128);
        k_zero<<<gNC, TPB, 0, stream>>>(agg, nNC);
        k_scatter<<<cdiv((long long)NEDGES * 32, TPB), TPB, 0, stream>>>(lin, ei, dis, agg);
        k_combine<<<gNC, TPB, 0, stream>>>(agg, lin, dis, lb[l], lres[l], lout[l]);
    }

    // heads: h3 lives in hA
    k_f2bf<<<gNC, TPB, 0, stream>>>(hA, hbf, nNC);

    // radius head: Linear(128->64) + BN + ReLU -> Linear(64->1) -> softplus
    k_wt_bf<<<cdiv(128 * 64, TPB), TPB, 0, stream>>>(rW1, wbf, 64);
    k_gemm_bf16<<<dim3((int)(N / 16)), dim3(4 * 32), 0, stream>>>(hbf, wbf, rbuf, 64);
    k_bn_relu<<<cdiv((long long)N * 64, TPB), TPB, 0, stream>>>(rbuf, rb1, rbn1g, rbn1b, rbn1m, rbn1v, (int)(N * 64), 6);
    k_radius<<<gN, TPB, 0, stream>>>(rbuf, rW2, rb2, radius);

    // position head: Linear(128->128)+BN+ReLU -> Linear(128->64)+BN+ReLU -> Linear(64->2)
    k_wt_bf<<<cdiv(128 * 128, TPB), TPB, 0, stream>>>(pW1, wbf, 128);
    k_gemm_bf16<<<dim3((int)(N / 16)), dim3(8 * 32), 0, stream>>>(hbf, wbf, lin, 128);
    k_bn_relu<<<gNC, TPB, 0, stream>>>(lin, pb1, pbn1g, pbn1b, pbn1m, pbn1v, nNC, 7);
    k_f2bf<<<gNC, TPB, 0, stream>>>(lin, hbf, nNC);
    k_wt_bf<<<cdiv(128 * 64, TPB), TPB, 0, stream>>>(pW2, wbf, 64);
    k_gemm_bf16<<<dim3((int)(N / 16)), dim3(4 * 32), 0, stream>>>(hbf, wbf, c2, 64);
    k_bn_relu<<<cdiv((long long)N * 64, TPB), TPB, 0, stream>>>(c2, pb2, pbn2g, pbn2b, pbn2m, pbn2v, (int)(N * 64), 6);

    k_final<<<gN, TPB, 0, stream>>>(c2, pW3, pb3, radius, (float*)d_out);
}
